// channelAttention_24232205484502
// MI455X (gfx1250) — matrix operand order repaired and checked by reading, not compiled, not run on hardware
//
#include <hip/hip_runtime.h>
#include <hip/hip_bf16.h>

// ---- sizes from the reference ----
#define BB   16
#define CC   512
#define CH   256
#define HW   16384      // 128*128
#define NBC  (BB*CC)    // 8192

typedef __attribute__((ext_vector_type(4)))  float    v4f;
typedef __attribute__((ext_vector_type(8)))  float    v8f;
typedef __attribute__((ext_vector_type(16))) _Float16 v16h;

// -----------------------------------------------------------------------------
// Kernel 1: fused global avg+max pool. One block per (b,c) slice (64 KB
// contiguous). This is the only bandwidth-relevant kernel: 536 MB / 23.3 TB/s
// ~= 23 us. Nontemporal b128 loads: one-pass stream, keep L2 clean.
// pools layout: pools[0*NBC + b*CC + c] = mean, pools[1*NBC + b*CC + c] = max.
// -----------------------------------------------------------------------------
__global__ __launch_bounds__(256) void pool_kernel(const float* __restrict__ x,
                                                   float* __restrict__ pools) {
    const int bc = blockIdx.x;
    const v4f* p4 = (const v4f*)(x + (size_t)bc * HW);

    float s  = 0.0f;
    float mx = -3.402823466e38f;
#pragma unroll
    for (int k = 0; k < 16; ++k) {
        v4f v = __builtin_nontemporal_load(p4 + threadIdx.x + k * 256);
        s  += (v.x + v.y) + (v.z + v.w);
        mx  = fmaxf(mx, fmaxf(fmaxf(v.x, v.y), fmaxf(v.z, v.w)));
    }
    // wave32 butterfly reduction
#pragma unroll
    for (int off = 16; off > 0; off >>= 1) {
        s  += __shfl_xor(s, off, 32);
        mx  = fmaxf(mx, __shfl_xor(mx, off, 32));
    }
    __shared__ float ws[8];
    __shared__ float wm[8];
    const int wave = threadIdx.x >> 5;
    const int lane = threadIdx.x & 31;
    if (lane == 0) { ws[wave] = s; wm[wave] = mx; }
    __syncthreads();
    if (threadIdx.x == 0) {
        float S = ws[0], M = wm[0];
#pragma unroll
        for (int w = 1; w < 8; ++w) { S += ws[w]; M = fmaxf(M, wm[w]); }
        pools[bc]       = S * (1.0f / (float)HW);
        pools[NBC + bc] = M;
    }
}

// -----------------------------------------------------------------------------
// Kernel 2: both excitation MLPs via WMMA (wave32, 16x16x32 f16 -> f32 acc).
//   h  = relu(P @ W1^T)        [16,512]x[512,256] -> [16,256]
//   y  = sigmoid(h @ W2^T)     [16,256]x[256,512] -> [16,512]
//   out = y_avg + y_max        (softmax over singleton axis == ones)
// Single 512-thread block (16 waves); work is trivially small vs the pool pass.
// -----------------------------------------------------------------------------
__global__ __launch_bounds__(512) void mlp_kernel(const float* __restrict__ pools,
                                                  const float* __restrict__ aw1,
                                                  const float* __restrict__ aw2,
                                                  const float* __restrict__ mw1,
                                                  const float* __restrict__ mw2,
                                                  float* __restrict__ out) {
    __shared__ _Float16 sP[2][BB][CC];   // pooled inputs (avg, max)  32 KB
    __shared__ _Float16 sH[2][BB][CH];   // relu hidden   (avg, max)  16 KB

    const int tid  = threadIdx.x;        // 0..511
    const int wave = tid >> 5;           // 0..15
    const int lane = tid & 31;
    const int ln16 = lane & 15;          // row (A) / col (B,C,D) within tile
    const int half = lane >> 4;          // lane-group select

    // pools (fp32) -> LDS fp16; flat layouts match ([branch][b][c])
    _Float16* sPf = &sP[0][0][0];
    for (int i = tid; i < 2 * BB * CC; i += 512) sPf[i] = (_Float16)pools[i];
    __syncthreads();

    // ---------------- Stage A: h = relu(P @ W1^T), 32 tiles (2 branches x 16)
    for (int t = wave * 2; t < wave * 2 + 2; ++t) {
        const int br = t >> 4;           // 0 = avg, 1 = max
        const int nt = t & 15;           // output-channel tile (of CH=256)
        const float* __restrict__ w1 = br ? mw1 : aw1;
        v8f acc = {};
        for (int kb = 0; kb < CC; kb += 32) {
            v16h a, b;
#pragma unroll
            for (int e = 0; e < 16; ++e) {
                // A 16x32 f16 layout: K = kb + 16*(e/8) + 8*half + (e%8)
                const int ka = kb + ((e >> 3) << 4) + (half << 3) + (e & 7);
                a[e] = sP[br][ln16][ka];
                // B 32x16 f16 layout: K = kb + 16*(e/8) + 8*half + (e%8) ; N = ln16
                const int kk = kb + ((e >> 3) << 4) + (half << 3) + (e & 7);
                b[e] = (_Float16)w1[(nt * 16 + ln16) * CC + kk];
            }
            acc = __builtin_amdgcn_wmma_f32_16x16x32_f16(
                false, a, false, b, (short)0, acc, false, false);
        }
        // D 16x16 f32: element r -> M = r + 8*half, N = ln16
#pragma unroll
        for (int r = 0; r < 8; ++r)
            sH[br][r + (half << 3)][nt * 16 + ln16] = (_Float16)fmaxf(acc[r], 0.0f);
    }
    __syncthreads();

    // ---------------- Stage B: out = sig(h_a @ W2a^T) + sig(h_m @ W2m^T)
    for (int nt = wave * 2; nt < wave * 2 + 2; ++nt) {   // 32 tiles of C=512
        float res[8];
#pragma unroll
        for (int r = 0; r < 8; ++r) res[r] = 0.0f;
        for (int br = 0; br < 2; ++br) {
            const float* __restrict__ w2 = br ? mw2 : aw2;
            v8f acc = {};
            for (int kb = 0; kb < CH; kb += 32) {
                v16h a, b;
#pragma unroll
                for (int e = 0; e < 16; ++e) {
                    const int ka = kb + ((e >> 3) << 4) + (half << 3) + (e & 7);
                    a[e] = sH[br][ln16][ka];
                    const int kk = kb + ((e >> 3) << 4) + (half << 3) + (e & 7);
                    b[e] = (_Float16)w2[(nt * 16 + ln16) * CH + kk];
                }
                acc = __builtin_amdgcn_wmma_f32_16x16x32_f16(
                    false, a, false, b, (short)0, acc, false, false);
            }
#pragma unroll
            for (int r = 0; r < 8; ++r)
                res[r] += 1.0f / (1.0f + __expf(-acc[r]));
        }
#pragma unroll
        for (int r = 0; r < 8; ++r)
            out[(r + (half << 3)) * CC + nt * 16 + ln16] = res[r];
    }
}

extern "C" void kernel_launch(void* const* d_in, const int* in_sizes, int n_in,
                              void* d_out, int out_size, void* d_ws, size_t ws_size,
                              hipStream_t stream) {
    const float* x      = (const float*)d_in[0];
    const float* avg_w1 = (const float*)d_in[1];
    const float* avg_w2 = (const float*)d_in[2];
    const float* max_w1 = (const float*)d_in[3];
    const float* max_w2 = (const float*)d_in[4];
    // d_in[5] (da_w1) and d_in[6] (da_w2) are dead: softmax over a singleton
    // axis yields all-ones attention, so out == avg_x + max_x exactly.
    (void)in_sizes; (void)n_in; (void)out_size; (void)ws_size;

    float* pools = (float*)d_ws;   // 2 * 8192 fp32 = 64 KB scratch

    pool_kernel<<<NBC, 256, 0, stream>>>(x, pools);
    mlp_kernel<<<1, 512, 0, stream>>>(pools, avg_w1, avg_w2, max_w1, max_w2,
                                      (float*)d_out);
}
